// Attention_40518721470947
// MI455X (gfx1250) — compile-verified, matrix operand order checked by reading, not run on hardware
//
#include <hip/hip_runtime.h>
#include <cstdint>

#define B_  8
#define TQ_ 2048
#define TK_ 2048
#define F_  1024
#define H_  1024

typedef __attribute__((ext_vector_type(16))) __bf16 v16bf;
typedef __attribute__((ext_vector_type(8)))  float  v8f;

union BFrag { v16bf v; unsigned short u[16]; uint4 q[2]; };
union CFrag { v8f v; float f[8]; };

#if __has_builtin(__builtin_amdgcn_global_load_async_to_lds_b128) && \
    __has_builtin(__builtin_amdgcn_s_wait_asynccnt)
#define HAVE_ASYNC_LDS 1
typedef int v4i_ __attribute__((vector_size(16)));
typedef __attribute__((address_space(1))) v4i_ as1_v4i;
typedef __attribute__((address_space(3))) v4i_ as3_v4i;
#else
#define HAVE_ASYNC_LDS 0
#endif

__device__ __forceinline__ unsigned short f2bf(float x) {
  union { float f; unsigned u; } c; c.f = x;
  unsigned r = c.u + 0x7fffu + ((c.u >> 16) & 1u);   // round-to-nearest-even
  return (unsigned short)(r >> 16);
}

// ---------------------------------------------------------------------------
// Kernel 1: C_bf16[M,H] = X_f32[M,F] @ W_f32[F,H] + bias   (M = B*T = 16384)
// 128x128 tile per workgroup, K-step 32, 8 waves each computing 32x64.
// TRANS=false: Out[row*H + col]            (row-major, for Q and K)
// TRANS=true : Out[(b*H + col)*TK + t]     (per-batch transposed, for V, so
//              the attention P.V B-operand is K-contiguous -> b128 loads)
// ---------------------------------------------------------------------------
template <bool TRANS>
__global__ __launch_bounds__(256) void proj_bf16_kernel(
    const float* __restrict__ X, const float* __restrict__ W,
    const float* __restrict__ bias, unsigned short* __restrict__ Out) {
  __shared__ unsigned short lA[128][40];   // [row][k]  (pad keeps 16B align)
  __shared__ unsigned short lB[128][40];   // [col][k]  (W tile transposed)
  const int tid  = threadIdx.x;
  const int lane = tid & 31, wave = tid >> 5;
  const int m0 = blockIdx.y << 7, n0 = blockIdx.x << 7;
  const int wm = (wave >> 1) * 32, wn = (wave & 1) * 64;
  const int ln = lane & 15;
  const int kb = (lane >> 4) << 3;

  CFrag acc[2][4];
  for (int i = 0; i < 2; ++i)
    for (int j = 0; j < 4; ++j)
      for (int e = 0; e < 8; ++e) acc[i][j].f[e] = 0.f;

  for (int k0 = 0; k0 < F_; k0 += 32) {
    // stage A tile: 128x32 f32 -> bf16
#pragma unroll
    for (int p = 0; p < 4; ++p) {
      const int q = tid + p * 256;
      const int row = q >> 3, c4 = (q & 7) << 2;
      const float4 xv = *(const float4*)&X[(size_t)(m0 + row) * F_ + k0 + c4];
      lA[row][c4 + 0] = f2bf(xv.x);
      lA[row][c4 + 1] = f2bf(xv.y);
      lA[row][c4 + 2] = f2bf(xv.z);
      lA[row][c4 + 3] = f2bf(xv.w);
    }
    // stage B tile transposed: W[k][n] -> lB[n][k]
#pragma unroll
    for (int p = 0; p < 4; ++p) {
      const int q = tid + p * 256;
      const int kr = q >> 5, c4 = (q & 31) << 2;
      const float4 wv = *(const float4*)&W[(size_t)(k0 + kr) * H_ + n0 + c4];
      lB[c4 + 0][kr] = f2bf(wv.x);
      lB[c4 + 1][kr] = f2bf(wv.y);
      lB[c4 + 2][kr] = f2bf(wv.z);
      lB[c4 + 3][kr] = f2bf(wv.w);
    }
    __syncthreads();
#pragma unroll
    for (int mi = 0; mi < 2; ++mi) {
      BFrag a;
      const int ar = wm + mi * 16 + ln;
      a.q[0] = *(const uint4*)&lA[ar][kb];
      a.q[1] = *(const uint4*)&lA[ar][kb + 16];
#pragma unroll
      for (int ni = 0; ni < 4; ++ni) {
        BFrag b;
        const int bc = wn + ni * 16 + ln;
        b.q[0] = *(const uint4*)&lB[bc][kb];
        b.q[1] = *(const uint4*)&lB[bc][kb + 16];
        acc[mi][ni].v = __builtin_amdgcn_wmma_f32_16x16x32_bf16(
            false, a.v, false, b.v, (short)0, acc[mi][ni].v, false, false);
      }
    }
    __syncthreads();
  }
  // fused bias + bf16 store
#pragma unroll
  for (int mi = 0; mi < 2; ++mi)
#pragma unroll
    for (int ni = 0; ni < 4; ++ni) {
      const int col = n0 + wn + ni * 16 + ln;
      const float bias_c = bias[col];
#pragma unroll
      for (int e = 0; e < 8; ++e) {
        const int row = m0 + wm + mi * 16 + e + ((lane >> 4) << 3);
        const unsigned short val = f2bf(acc[mi][ni].f[e] + bias_c);
        if (!TRANS) {
          Out[(size_t)row * H_ + col] = val;
        } else {
          const int bi = row >> 11;            // row / TK_
          const int t  = row & (TK_ - 1);      // row % TK_
          Out[((size_t)bi * H_ + col) * TK_ + t] = val;
        }
      }
    }
}

// ---------------------------------------------------------------------------
// Kernel 2: flash-attention. One WG per (batch, 16 q-rows); 8 waves, each
// owning a 128-wide slice of H for Q.K^T partials and the O accumulator.
// K is row-major [b][key][h]; V is transposed [b][h][key], so both the
// S-stage and PV-stage B-fragments are contiguous 16B global loads.
// The boolean mask tile is staged into LDS with the async-copy engine
// (GLOBAL_LOAD_ASYNC_TO_LDS_B128, overlapped with the S-stage WMMAs).
// ---------------------------------------------------------------------------
__global__ __launch_bounds__(256) void flash_attn_kernel(
    const unsigned short* __restrict__ Qb, const unsigned short* __restrict__ Kb,
    const unsigned short* __restrict__ Vt, const unsigned char* __restrict__ mask,
    float* __restrict__ out) {
  __shared__ float sS[16][64];            // fp32 score tile (cross-wave reduce)
  __shared__ unsigned short sP[16][72];   // bf16 probabilities
  __shared__ uint4 sMask[16][4];          // 16 rows x 64 mask bytes
  __shared__ float sM[16], sL[16], sAlpha[16];

  const int tid = threadIdx.x, lane = tid & 31, wave = tid >> 5;
  const int q0 = blockIdx.x << 4;         // 16 query rows
  const int b  = blockIdx.y;
  const int hb = wave << 7;               // this wave's H slice
  const int ln = lane & 15;
  const int kb = (lane >> 4) << 3;
  const size_t qbase  = ((size_t)b * TQ_ + q0) * H_;
  const size_t kbase  = (size_t)b * TK_ * (size_t)H_;   // K rows base
  const size_t vtbase = (size_t)b * H_ * (size_t)TK_;   // Vt rows base

  if (tid < 16) { sM[tid] = -__builtin_inff(); sL[tid] = 0.f; }

  // preload Q fragments for this wave's H slice (4 K-chunks of 32)
  BFrag qf[4];
#pragma unroll
  for (int kc = 0; kc < 4; ++kc) {
    const unsigned short* p = Qb + qbase + (size_t)ln * H_ + hb + kc * 32 + kb;
    qf[kc].q[0] = *(const uint4*)p;
    qf[kc].q[1] = *(const uint4*)(p + 16);
  }
  CFrag o[8];
  for (int ni = 0; ni < 8; ++ni)
    for (int e = 0; e < 8; ++e) o[ni].f[e] = 0.f;
  __syncthreads();

  for (int kt = 0; kt < TK_; kt += 64) {
#if HAVE_ASYNC_LDS
    // async-stage the 16x64B mask tile: wave 0, 2x16B per lane
    if (wave == 0) {
      const int r  = lane >> 1;
      const int co = (lane & 1) * 32;
      const unsigned char* gp = mask + ((size_t)b * TQ_ + q0 + r) * TK_ + kt + co;
      unsigned char* lp = (unsigned char*)&sMask[r][0] + co;
      as1_v4i* g = (as1_v4i*)(uintptr_t)gp;
      as3_v4i* l = (as3_v4i*)(unsigned)(uintptr_t)lp;
      __builtin_amdgcn_global_load_async_to_lds_b128(g, l, 0, 0);
      __builtin_amdgcn_global_load_async_to_lds_b128(g, l, 16, 0);
    }
#endif
    if (kt + 64 < TK_) {      // warm next K/V tiles (global_prefetch_b8)
      __builtin_prefetch(Kb + kbase + (size_t)(kt + 64 + ln) * H_ + hb, 0, 1);
      __builtin_prefetch(Vt + vtbase + (size_t)(hb + lane * 4) * TK_ + kt + 64, 0, 1);
    }
    // zero score tile
    ((float4*)sS)[tid] = make_float4(0.f, 0.f, 0.f, 0.f);
    __syncthreads();

    // partial S (this wave's 128-dim slice of the H reduction)
    CFrag sfr[4];
#pragma unroll
    for (int ni = 0; ni < 4; ++ni)
#pragma unroll
      for (int e = 0; e < 8; ++e) sfr[ni].f[e] = 0.f;
#pragma unroll
    for (int kc = 0; kc < 4; ++kc) {
#pragma unroll
      for (int ni = 0; ni < 4; ++ni) {
        BFrag bk_;
        const unsigned short* p =
            Kb + kbase + (size_t)(kt + ni * 16 + ln) * H_ + hb + kc * 32 + kb;
        bk_.q[0] = *(const uint4*)p;
        bk_.q[1] = *(const uint4*)(p + 16);
        sfr[ni].v = __builtin_amdgcn_wmma_f32_16x16x32_bf16(
            false, qf[kc].v, false, bk_.v, (short)0, sfr[ni].v, false, false);
      }
    }
    // cross-wave reduction via LDS float atomics (ds_add_f32)
#pragma unroll
    for (int ni = 0; ni < 4; ++ni)
#pragma unroll
      for (int e = 0; e < 8; ++e)
        atomicAdd(&sS[e + ((lane >> 4) << 3)][ni * 16 + ln], sfr[ni].f[e]);
#if HAVE_ASYNC_LDS
    if (wave == 0) __builtin_amdgcn_s_wait_asynccnt(0);
#endif
    __syncthreads();

    // masked online softmax, one lane per row
    if (tid < 16) {
      const int r = tid;
#if HAVE_ASYNC_LDS
      const unsigned char* mrow = (const unsigned char*)&sMask[r][0];
#else
      const unsigned char* mrow = mask + ((size_t)b * TQ_ + q0 + r) * TK_ + kt;
#endif
      const float mOld = sM[r];
      float mNew = mOld;
#pragma unroll 8
      for (int c = 0; c < 64; ++c) {
        const float s = mrow[c] ? sS[r][c] * 0.03125f : -__builtin_inff();
        mNew = fmaxf(mNew, s);
      }
      const float alpha = (mNew == -__builtin_inff()) ? 1.f : __expf(mOld - mNew);
      float lsum = 0.f;
#pragma unroll 8
      for (int c = 0; c < 64; ++c) {
        const float s = mrow[c] ? sS[r][c] * 0.03125f : -__builtin_inff();
        const float pv = (s == -__builtin_inff()) ? 0.f : __expf(s - mNew);
        sP[r][c] = f2bf(pv);
        lsum += pv;
      }
      sM[r] = mNew;
      sL[r] = sL[r] * alpha + lsum;
      sAlpha[r] = alpha;
    }
    __syncthreads();

    // rescale running O by alpha
    float al[8];
#pragma unroll
    for (int e = 0; e < 8; ++e) al[e] = sAlpha[e + ((lane >> 4) << 3)];
#pragma unroll
    for (int ni = 0; ni < 8; ++ni)
#pragma unroll
      for (int e = 0; e < 8; ++e) o[ni].f[e] *= al[e];

    // O += P (16x64 bf16, LDS) x V_slice (64x128 bf16, from transposed Vt)
    BFrag pa[2];
#pragma unroll
    for (int ks = 0; ks < 2; ++ks) {
      pa[ks].q[0] = *(const uint4*)&sP[ln][ks * 32 + kb];
      pa[ks].q[1] = *(const uint4*)&sP[ln][ks * 32 + kb + 16];
    }
#pragma unroll
    for (int ni = 0; ni < 8; ++ni) {
      const int col_h = hb + ni * 16 + ln;
#pragma unroll
      for (int ks = 0; ks < 2; ++ks) {
        BFrag bv_;
        const unsigned short* p =
            Vt + vtbase + (size_t)col_h * TK_ + kt + ks * 32 + kb;
        bv_.q[0] = *(const uint4*)p;
        bv_.q[1] = *(const uint4*)(p + 16);
        o[ni].v = __builtin_amdgcn_wmma_f32_16x16x32_bf16(
            false, pa[ks].v, false, bv_.v, (short)0, o[ni].v, false, false);
      }
    }
    __syncthreads();
  }

  // normalize by l and store fp32
  float rl[8];
#pragma unroll
  for (int e = 0; e < 8; ++e) {
    const float l = sL[e + ((lane >> 4) << 3)];
    rl[e] = (l > 0.f) ? 1.f / l : 0.f;
  }
#pragma unroll
  for (int ni = 0; ni < 8; ++ni)
#pragma unroll
    for (int e = 0; e < 8; ++e) {
      const int row = q0 + e + ((lane >> 4) << 3);
      out[((size_t)b * TQ_ + row) * H_ + hb + ni * 16 + ln] = o[ni].f[e] * rl[e];
    }
}

// ---------------------------------------------------------------------------
extern "C" void kernel_launch(void* const* d_in, const int* in_sizes, int n_in,
                              void* d_out, int out_size, void* d_ws, size_t ws_size,
                              hipStream_t stream) {
  (void)in_sizes; (void)n_in; (void)out_size; (void)ws_size;
  const float* query        = (const float*)d_in[0];
  const float* enc          = (const float*)d_in[1];
  const unsigned char* mask = (const unsigned char*)d_in[2];
  const float* Wq = (const float*)d_in[3];
  const float* bq = (const float*)d_in[4];
  const float* Wk = (const float*)d_in[5];
  const float* bk = (const float*)d_in[6];
  const float* Wv = (const float*)d_in[7];
  const float* bv = (const float*)d_in[8];
  float* out = (float*)d_out;

  unsigned short* ws = (unsigned short*)d_ws;
  const size_t E = (size_t)B_ * TQ_ * H_;      // 16.78M elements (bf16)
  unsigned short* Qb = ws;
  unsigned short* Kb = ws + E;
  unsigned short* Vt = ws + 2 * E;             // stored [b][h][key]

  dim3 pg(H_ / 128, (B_ * TQ_) / 128);         // 8 x 128 tiles
  proj_bf16_kernel<false><<<pg, 256, 0, stream>>>(query, Wq, bq, Qb);
  proj_bf16_kernel<false><<<pg, 256, 0, stream>>>(enc,   Wk, bk, Kb);
  proj_bf16_kernel<true ><<<pg, 256, 0, stream>>>(enc,   Wv, bv, Vt);

  dim3 ag(TQ_ / 16, B_);                       // 128 x 8 workgroups
  flash_attn_kernel<<<ag, 256, 0, stream>>>(Qb, Kb, Vt, mask, out);
}
